// MultiHeadAttention_2534030705246
// MI455X (gfx1250) — compile-verified, matrix operand order checked by reading, not run on hardware
//
#include <hip/hip_runtime.h>
#include <hip/hip_bf16.h>

#define NTOK 4096
#define DMODEL 1024
#define NHEAD 16
#define HDIM 64

typedef __attribute__((ext_vector_type(16))) _Float16 v16h;
typedef __attribute__((ext_vector_type(8)))  _Float16 v8h;
typedef __attribute__((ext_vector_type(8)))  float    v8f;

// ---------------------------------------------------------------------------
// WMMA fragment loader (A-matrix 16x32 f16 layout per ISA 7.12.2):
// lane L holds row M=L%16; lanes 0-15 hold K in {0..7,16..23}, lanes 16-31
// hold K in {8..15,24..31}.  Same pattern used for B-fragments read from
// TRANSPOSED storage so each lane does two contiguous 16-byte loads.
// ---------------------------------------------------------------------------
template <typename PtrT>
__device__ __forceinline__ v16h load_frag(PtrT base, int ld) {
  const int lane = threadIdx.x & 31;
  const int row  = lane & 15;
  const int off  = (lane < 16) ? 0 : 8;
  PtrT p = base + (size_t)row * ld + off;
  v8h lo = *(const v8h*)(p);
  v8h hi = *(const v8h*)(p + 16);
  v16h r;
#pragma unroll
  for (int i = 0; i < 8; ++i) { r[i] = lo[i]; r[8 + i] = hi[i]; }
  return r;
}

__device__ __forceinline__ v8f wmma_f16(v16h a, v16h b, v8f c) {
  return __builtin_amdgcn_wmma_f32_16x16x32_f16(false, a, false, b,
                                                (short)0, c, false, false);
}

// CDNA5 async global->LDS copy (16B), tracked by ASYNCcnt.
__device__ __forceinline__ void async_copy_b128(const _Float16* gsrc,
                                                _Float16* ldst) {
  unsigned lds_off = (unsigned)(uintptr_t)ldst;          // low 32b = LDS addr
  unsigned long long ga = (unsigned long long)(uintptr_t)gsrc;
  asm volatile("global_load_async_to_lds_b128 %0, %1, off"
               :: "v"(lds_off), "v"(ga) : "memory");
}

__device__ __forceinline__ void wait_async_all() {
  asm volatile("s_wait_asynccnt 0x0" ::: "memory");
}

// ---------------------------------------------------------------------------
// Kernel 0: fp32 -> f16 conversion; weights stored transposed.
// ---------------------------------------------------------------------------
__global__ void mha_convert(const float* __restrict__ x,
                            const float* __restrict__ Wq,
                            const float* __restrict__ Wk,
                            const float* __restrict__ Wv,
                            const float* __restrict__ Wo,
                            _Float16* __restrict__ xh,
                            _Float16* __restrict__ wqt,
                            _Float16* __restrict__ wkt,
                            _Float16* __restrict__ wvt,
                            _Float16* __restrict__ wot) {
  size_t i = (size_t)blockIdx.x * blockDim.x + threadIdx.x;
  if (i < (size_t)NTOK * DMODEL) xh[i] = (_Float16)x[i];
  if (i < (size_t)DMODEL * DMODEL) {
    int d = (int)(i >> 10);
    int e = (int)(i & 1023);
    size_t t = (size_t)e * DMODEL + d;
    wqt[t] = (_Float16)Wq[i];
    wkt[t] = (_Float16)Wk[i];
    wvt[t] = (_Float16)Wv[i];
    wot[t] = (_Float16)Wo[i];
  }
}

// ---------------------------------------------------------------------------
// Kernel 1: QKV projections.  grid = (N/128, D/64, 3), block = 256 (8 waves).
// Wave w: rows [bx*128 + w*16, +16) x cols [by*64, +64).
// Software-pipelined: fragments for step k+32 are in flight while WMMAs for
// step k execute.  Q,K stored [H][N][HD]; V stored transposed [H][HD][N].
// ---------------------------------------------------------------------------
__global__ __launch_bounds__(256) void mha_proj(
    const _Float16* __restrict__ xh,
    const _Float16* __restrict__ wqt,
    const _Float16* __restrict__ wkt,
    const _Float16* __restrict__ wvt,
    _Float16* __restrict__ qh,
    _Float16* __restrict__ kh,
    _Float16* __restrict__ vth) {
  const int wave = threadIdx.x >> 5;
  const int lane = threadIdx.x & 31;
  const int mbase = blockIdx.x * 128 + wave * 16;
  const int nbase = blockIdx.y * 64;
  const int which = blockIdx.z;
  const _Float16* wt = (which == 0) ? wqt : (which == 1) ? wkt : wvt;
  const _Float16* ap = xh + (size_t)mbase * DMODEL;

  v8f acc[4];
#pragma unroll
  for (int ct = 0; ct < 4; ++ct) acc[ct] = (v8f)(0.0f);

  v16h a_cur = load_frag(ap, DMODEL);
  v16h b_cur[4];
#pragma unroll
  for (int ct = 0; ct < 4; ++ct)
    b_cur[ct] = load_frag(wt + (size_t)(nbase + ct * 16) * DMODEL, DMODEL);

#pragma unroll 2
  for (int k = 0; k < DMODEL; k += 32) {
    const int kn = (k + 32 < DMODEL) ? k + 32 : k;   // dummy reload on last
    v16h a_nxt = load_frag(ap + kn, DMODEL);
    v16h b_nxt[4];
#pragma unroll
    for (int ct = 0; ct < 4; ++ct)
      b_nxt[ct] = load_frag(wt + (size_t)(nbase + ct * 16) * DMODEL + kn, DMODEL);
#pragma unroll
    for (int ct = 0; ct < 4; ++ct)
      acc[ct] = wmma_f16(a_cur, b_cur[ct], acc[ct]);
    a_cur = a_nxt;
#pragma unroll
    for (int ct = 0; ct < 4; ++ct) b_cur[ct] = b_nxt[ct];
  }

  const int rowoff = (lane >> 4) << 3;
  const int col0   = lane & 15;
#pragma unroll
  for (int ct = 0; ct < 4; ++ct) {
#pragma unroll
    for (int v = 0; v < 8; ++v) {
      int n = mbase + v + rowoff;
      int e = nbase + ct * 16 + col0;
      int h  = e >> 6;
      int hd = e & 63;
      _Float16 val = (_Float16)acc[ct][v];
      if (which == 0)      qh[((size_t)h * NTOK + n) * HDIM + hd] = val;
      else if (which == 1) kh[((size_t)h * NTOK + n) * HDIM + hd] = val;
      else                 vth[((size_t)h * HDIM + hd) * NTOK + n] = val;
    }
  }
}

// ---------------------------------------------------------------------------
// Kernel 2: fused causal flash attention.
// grid = (N/64, H), block = 128 (4 waves); wave w owns 16 query rows.
// K/V tiles staged in LDS via async global->LDS copies, double-buffered
// across the key-tile loop.  P staged via per-wave LDS (DS ops in-order
// within a wave -> no barrier needed for P).
// ---------------------------------------------------------------------------
__global__ __launch_bounds__(128) void mha_attn(
    const _Float16* __restrict__ qh,
    const _Float16* __restrict__ kh,
    const _Float16* __restrict__ vth,
    _Float16* __restrict__ ctxh) {
  const int h    = blockIdx.y;
  const int qt   = blockIdx.x;
  const int tid  = threadIdx.x;
  const int wave = tid >> 5;
  const int lane = tid & 31;
  const int qbase = qt * 64 + wave * 16;
  const int rowoff = (lane >> 4) << 3;
  const int col0   = lane & 15;
  const float L2E = 1.4426950408889634f;

  __shared__ __align__(16) _Float16 Kbuf[2][64][64];   // 16 KB
  __shared__ __align__(16) _Float16 Vbuf[2][64][64];   // 16 KB (rows = hd)
  __shared__ __align__(16) _Float16 Pbuf[4][16][64];   // 8 KB per-wave strips

  const _Float16* kheadp = kh  + (size_t)h * NTOK * HDIM;
  const _Float16* vheadp = vth + (size_t)h * HDIM * NTOK;

  // stage key-tile kt into buffer b: 64x64 K rows + 64x64 V^T rows
  auto stage = [&](int kt, int b) {
    const int kbase = kt * 64;
    const _Float16* ks = kheadp + (size_t)kbase * HDIM;   // row=key, ld=64
    const _Float16* vs = vheadp + kbase;                  // row=hd,  ld=N
#pragma unroll
    for (int i = 0; i < 4; ++i) {
      int cid = tid * 4 + i;            // 512 chunks of 16B
      int row = cid >> 3;
      int c   = (cid & 7) * 8;
      async_copy_b128(ks + (size_t)row * HDIM + c, &Kbuf[b][row][c]);
      async_copy_b128(vs + (size_t)row * NTOK + c, &Vbuf[b][row][c]);
    }
  };

  const _Float16* qptr = qh + ((size_t)h * NTOK + qbase) * HDIM;
  v16h qf0 = load_frag(qptr, HDIM);
  v16h qf1 = load_frag(qptr + 32, HDIM);

  v8f ctx[4];
#pragma unroll
  for (int dt = 0; dt < 4; ++dt) ctx[dt] = (v8f)(0.0f);
  float mst[8], lst[8];
#pragma unroll
  for (int v = 0; v < 8; ++v) { mst[v] = -__builtin_inff(); lst[v] = 0.0f; }

  stage(0, 0);
  wait_async_all();
  __syncthreads();

  for (int kt = 0; kt <= qt; ++kt) {
    const int cur = kt & 1;
    if (kt < qt) stage(kt + 1, cur ^ 1);   // prefetch next tile (async)

    // ---- S = (Q . K^T) / 8 for this wave's 16x64 strip -------------------
    v8f s[4];
#pragma unroll
    for (int ct = 0; ct < 4; ++ct) {
      const _Float16* kp = &Kbuf[cur][ct * 16][0];
      v16h b0 = load_frag(kp, 64);
      v16h b1 = load_frag(kp + 32, 64);
      v8f a = (v8f)(0.0f);
      a = wmma_f16(qf0, b0, a);
      a = wmma_f16(qf1, b1, a);
#pragma unroll
      for (int v = 0; v < 8; ++v) a[v] *= 0.125f;   // 1/sqrt(64)
      s[ct] = a;
    }

    // ---- causal mask on diagonal tile ------------------------------------
    if (kt == qt) {
#pragma unroll
      for (int ct = 0; ct < 4; ++ct)
#pragma unroll
        for (int v = 0; v < 8; ++v) {
          int key = ct * 16 + col0;
          int qr  = wave * 16 + v + rowoff;
          if (key > qr) s[ct][v] = -__builtin_inff();
        }
    }

    // ---- online softmax ---------------------------------------------------
    float corr[8];
#pragma unroll
    for (int v = 0; v < 8; ++v) {
      float mx = fmaxf(fmaxf(s[0][v], s[1][v]), fmaxf(s[2][v], s[3][v]));
      mx = fmaxf(mx, __shfl_xor(mx, 1));
      mx = fmaxf(mx, __shfl_xor(mx, 2));
      mx = fmaxf(mx, __shfl_xor(mx, 4));
      mx = fmaxf(mx, __shfl_xor(mx, 8));
      float mnew = fmaxf(mst[v], mx);
      corr[v] = __builtin_exp2f((mst[v] - mnew) * L2E);
      mst[v] = mnew;

      float lsum = 0.0f;
#pragma unroll
      for (int ct = 0; ct < 4; ++ct) {
        float p = __builtin_exp2f((s[ct][v] - mnew) * L2E);
        lsum += p;
        Pbuf[wave][v + rowoff][ct * 16 + col0] = (_Float16)p;
      }
      lsum += __shfl_xor(lsum, 1);
      lsum += __shfl_xor(lsum, 2);
      lsum += __shfl_xor(lsum, 4);
      lsum += __shfl_xor(lsum, 8);
      lst[v] = lst[v] * corr[v] + lsum;
    }
#pragma unroll
    for (int dt = 0; dt < 4; ++dt)
#pragma unroll
      for (int v = 0; v < 8; ++v) ctx[dt][v] *= corr[v];

    // ---- ctx += P . V ----------------------------------------------------
    const _Float16* pb = &Pbuf[wave][0][0];
    v16h pf0 = load_frag(pb, 64);        // keys 0..31 of tile
    v16h pf1 = load_frag(pb + 32, 64);   // keys 32..63
#pragma unroll
    for (int dt = 0; dt < 4; ++dt) {
      const _Float16* vp = &Vbuf[cur][dt * 16][0];
      v16h vb0 = load_frag(vp, 64);
      v16h vb1 = load_frag(vp + 32, 64);
      ctx[dt] = wmma_f16(pf0, vb0, ctx[dt]);
      ctx[dt] = wmma_f16(pf1, vb1, ctx[dt]);
    }

    // next buffer must be resident; all waves done reading current buffer
    wait_async_all();
    __syncthreads();
  }

  // ---- normalize and store ctx as [N][D] f16 ------------------------------
#pragma unroll
  for (int v = 0; v < 8; ++v) {
    float inv = 1.0f / lst[v];
    int n = qbase + v + rowoff;
#pragma unroll
    for (int dt = 0; dt < 4; ++dt) {
      int e = h * HDIM + dt * 16 + col0;
      ctxh[(size_t)n * DMODEL + e] = (_Float16)(ctx[dt][v] * inv);
    }
  }
}

// ---------------------------------------------------------------------------
// Kernel 3: output projection ctx . Wo + bo -> fp32, software-pipelined.
// grid = (N/128, D/64), block = 256 (8 waves).
// ---------------------------------------------------------------------------
__global__ __launch_bounds__(256) void mha_oproj(
    const _Float16* __restrict__ ctxh,
    const _Float16* __restrict__ wot,
    const float* __restrict__ bo,
    float* __restrict__ out) {
  const int wave = threadIdx.x >> 5;
  const int lane = threadIdx.x & 31;
  const int mbase = blockIdx.x * 128 + wave * 16;
  const int nbase = blockIdx.y * 64;
  const _Float16* ap = ctxh + (size_t)mbase * DMODEL;

  v8f acc[4];
#pragma unroll
  for (int ct = 0; ct < 4; ++ct) acc[ct] = (v8f)(0.0f);

  v16h a_cur = load_frag(ap, DMODEL);
  v16h b_cur[4];
#pragma unroll
  for (int ct = 0; ct < 4; ++ct)
    b_cur[ct] = load_frag(wot + (size_t)(nbase + ct * 16) * DMODEL, DMODEL);

#pragma unroll 2
  for (int k = 0; k < DMODEL; k += 32) {
    const int kn = (k + 32 < DMODEL) ? k + 32 : k;
    v16h a_nxt = load_frag(ap + kn, DMODEL);
    v16h b_nxt[4];
#pragma unroll
    for (int ct = 0; ct < 4; ++ct)
      b_nxt[ct] = load_frag(wot + (size_t)(nbase + ct * 16) * DMODEL + kn, DMODEL);
#pragma unroll
    for (int ct = 0; ct < 4; ++ct)
      acc[ct] = wmma_f16(a_cur, b_cur[ct], acc[ct]);
    a_cur = a_nxt;
#pragma unroll
    for (int ct = 0; ct < 4; ++ct) b_cur[ct] = b_nxt[ct];
  }

  const int rowoff = (lane >> 4) << 3;
  const int col0   = lane & 15;
#pragma unroll
  for (int ct = 0; ct < 4; ++ct) {
    int e = nbase + ct * 16 + col0;
    float bias = bo[e];
#pragma unroll
    for (int v = 0; v < 8; ++v) {
      int n = mbase + v + rowoff;
      out[(size_t)n * DMODEL + e] = acc[ct][v] + bias;
    }
  }
}

// ---------------------------------------------------------------------------
extern "C" void kernel_launch(void* const* d_in, const int* in_sizes, int n_in,
                              void* d_out, int out_size, void* d_ws, size_t ws_size,
                              hipStream_t stream) {
  const float* x  = (const float*)d_in[0];
  const float* Wq = (const float*)d_in[1];
  const float* Wk = (const float*)d_in[2];
  const float* Wv = (const float*)d_in[3];
  const float* Wo = (const float*)d_in[4];
  const float* bo = (const float*)d_in[5];
  float* out = (float*)d_out;

  const size_t MD = (size_t)NTOK * DMODEL;      // 4M halves
  const size_t WW = (size_t)DMODEL * DMODEL;    // 1M halves
  _Float16* ws   = (_Float16*)d_ws;
  _Float16* xh   = ws;
  _Float16* wqt  = xh  + MD;
  _Float16* wkt  = wqt + WW;
  _Float16* wvt  = wkt + WW;
  _Float16* wot  = wvt + WW;
  _Float16* qh   = wot + WW;          // [H][N][HD]
  _Float16* kh   = qh  + MD;          // [H][N][HD]
  _Float16* vth  = kh  + MD;          // [H][HD][N]
  _Float16* ctxh = vth + MD;          // [N][D]
  // total 24M halves = 48 MB of d_ws

  (void)in_sizes; (void)n_in; (void)out_size; (void)ws_size;

  mha_convert<<<(unsigned)((MD + 255) / 256), 256, 0, stream>>>(
      x, Wq, Wk, Wv, Wo, xh, wqt, wkt, wvt, wot);

  dim3 gp(NTOK / 128, DMODEL / 64, 3);
  mha_proj<<<gp, 256, 0, stream>>>(xh, wqt, wkt, wvt, qh, kh, vth);

  dim3 ga(NTOK / 64, NHEAD);
  mha_attn<<<ga, 128, 0, stream>>>(qh, kh, vth, ctxh);

  dim3 go(NTOK / 128, DMODEL / 64);
  mha_oproj<<<go, 256, 0, stream>>>(ctxh, wot, bo, out);
}
